// attentiona_51213190037761
// MI455X (gfx1250) — matrix operand order repaired and checked by reading, not compiled, not run on hardware
//
#include <hip/hip_runtime.h>
#include <hip/hip_bf16.h>
#include <math.h>

#define DEV __device__ __forceinline__

typedef __attribute__((ext_vector_type(16))) _Float16 v16h;
typedef __attribute__((ext_vector_type(8)))  float    v8f;
typedef __attribute__((ext_vector_type(4)))  unsigned int u32x4;
typedef __attribute__((ext_vector_type(4)))  int          i32x4;
typedef __attribute__((ext_vector_type(8)))  int          i32x8;

constexpr int B_   = 2;
constexpr int S_   = 2048;
constexpr int D_   = 512;
constexpr int H_   = 8;
constexpr int DH_  = 64;
constexpr int NT_  = B_ * S_;       // 4096 token rows
constexpr float LN_EPS_ = 1e-5f;
constexpr float SCALEF_ = 0.125f;   // HEAD_DIM^-0.5

DEV int lane_id() { return (int)(threadIdx.x & 31); }

DEV v8f wmma16(v16h a, v16h b, v8f c) {
  // v_wmma_f32_16x16x32_f16: D = A(16x32 f16) x B(32x16 f16) + C(16x16 f32)
  return __builtin_amdgcn_wmma_f32_16x16x32_f16(false, a, false, b, (short)0, c,
                                                false, false);
}

// ---- WMMA fragment loaders (wave32; contiguous k order, A and B match) ----
// A 16x32 f16: lanes 0-15 -> M=lane, K in {0..15}; lanes 16-31 -> M=lane-16,
// K in {16..31}. s points at tile corner (m0,k0), row-major pitch ld halves.
DEV v16h load_a(const _Float16* s, int ld) {
  int l = lane_id();
  int m = l & 15;
  int hb = (l >> 4) ? 8 : 0;
  v16h a;
#pragma unroll
  for (int i = 0; i < 16; ++i) {
    int g = i >> 1, r = i & 1;
    int k = 2 * g + 2 * hb + r;
    a[i] = s[m * ld + k];
  }
  return a;
}

// B 32x16 f16 from an [n][k]-major tile (outer index = column n, inner = k).
DEV v16h load_b_nk(const _Float16* s, int ld) {
  int l = lane_id();
  int n = l & 15, kh = (l >> 4) * 16;
  v16h b;
#pragma unroll
  for (int i = 0; i < 16; ++i) b[i] = s[n * ld + kh + i];
  return b;
}

// B 32x16 f16 from a [k][n]-major tile (outer index = k, inner = column n).
DEV v16h load_b_kn(const _Float16* s, int ld) {
  int l = lane_id();
  int n = l & 15, kh = (l >> 4) * 16;
  v16h b;
#pragma unroll
  for (int i = 0; i < 16; ++i) b[i] = s[(kh + i) * ld + n];
  return b;
}

DEV float wave_allred(float v) {
#pragma unroll
  for (int m = 16; m >= 1; m >>= 1) v += __shfl_xor(v, m, 32);
  return v;
}

// ---- Tensor Data Mover: DMA one [64 rows x 64 f16] tile (row stride 64) from
// global into LDS, depositing with 8-half padding per row (pad 4 DWORDs every
// 32 DWORDs) so the LDS pitch is 72 halves. Issued once per wave; TENSORcnt.
DEV void tdm_load_tile64(const _Float16* gsrc, void* ldst) {
  unsigned long long ga = (unsigned long long)(uintptr_t)gsrc;
  unsigned int la = (unsigned int)(uintptr_t)ldst;  // LDS byte offset (low 32b)
  u32x4 g0;
  g0[0] = 1u;                                  // count=1, user descriptor
  g0[1] = la;                                  // lds_addr
  g0[2] = (unsigned int)(ga & 0xFFFFFFFFu);    // global_addr[31:0]
  g0[3] = (unsigned int)((ga >> 32) & 0x01FFFFFFu) | (2u << 30);  // addr[56:32], type=2
  i32x8 g1;
  g1[0] = (int)((1u << 16)     // data_size = 2 bytes
              | (1u << 20)     // pad_enable
              | (4u << 22)     // pad_interval: every 32 DWORDs
              | (3u << 25));   // pad_amount: 4 DWORDs (8 halves) -> pitch 72
  g1[1] = (int)(64u << 16);    // tensor_dim0 = 64 (low 16 in [31:16])
  g1[2] = (int)(2048u << 16);  // tensor_dim0 hi | tensor_dim1 = 2048 low
  g1[3] = (int)(64u << 16);    // tensor_dim1 hi | tile_dim0 = 64
  g1[4] = 64;                  // tile_dim1 = 64, tile_dim2 = 0
  g1[5] = 64;                  // tensor_dim0_stride low 32 = 64
  g1[6] = 0;                   // stride hi | tensor_dim1_stride low
  g1[7] = 0;
  i32x4 z4 = {0, 0, 0, 0};
#if __clang_major__ >= 23
  i32x8 z8 = {0, 0, 0, 0, 0, 0, 0, 0};
  __builtin_amdgcn_tensor_load_to_lds(g0, g1, z4, z4, z8, 0);
#else
  __builtin_amdgcn_tensor_load_to_lds(g0, g1, z4, z4, 0);
#endif
}

// ---------------- Kernel 1: row LN_a + f16 casts ----------------
__global__ __launch_bounds__(256) void prep_x_kernel(
    const float* __restrict__ x, const float* __restrict__ lw,
    const float* __restrict__ lb, _Float16* __restrict__ x16,
    _Float16* __restrict__ xln16) {
  int row = blockIdx.x;                // 0..NT_-1
  int tid = threadIdx.x;
  const float* xr = x + (size_t)row * D_;
  __shared__ float red[256];
  float v0 = xr[tid], v1 = xr[tid + 256];
  red[tid] = v0 + v1;
  __syncthreads();
#pragma unroll
  for (int s = 128; s > 0; s >>= 1) {
    if (tid < s) red[tid] += red[tid + s];
    __syncthreads();
  }
  float mean = red[0] * (1.f / (float)D_);
  __syncthreads();
  float d0 = v0 - mean, d1 = v1 - mean;
  red[tid] = d0 * d0 + d1 * d1;
  __syncthreads();
#pragma unroll
  for (int s = 128; s > 0; s >>= 1) {
    if (tid < s) red[tid] += red[tid + s];
    __syncthreads();
  }
  float rstd = rsqrtf(red[0] * (1.f / (float)D_) + LN_EPS_);
  size_t o = (size_t)row * D_;
  x16[o + tid]         = (_Float16)v0;
  x16[o + tid + 256]   = (_Float16)v1;
  xln16[o + tid]       = (_Float16)(d0 * rstd * lw[tid] + lb[tid]);
  xln16[o + tid + 256] = (_Float16)(d1 * rstd * lw[tid + 256] + lb[tid + 256]);
}

// ---------------- Kernel 2: f32 -> f16 convert ----------------
__global__ __launch_bounds__(256) void cvt_f16_kernel(
    const float* __restrict__ a, _Float16* __restrict__ o, int n) {
  int i = blockIdx.x * blockDim.x + threadIdx.x;
  int st = gridDim.x * blockDim.x;
  for (; i < n; i += st) o[i] = (_Float16)a[i];
}

// ---------------- Kernel 3: C[rows x N] = A @ W^T + bias (f16 in, f32 out) ---
__global__ __launch_bounds__(256) void gemm_wt_kernel(
    const _Float16* __restrict__ A, const _Float16* __restrict__ W,
    const float* __restrict__ bias, float* __restrict__ C, int N, int K) {
  __shared__ _Float16 sA[64][40];
  __shared__ _Float16 sB[64][40];
  int tid = threadIdx.x;
  int wave = tid >> 5;
  int wm = wave & 3;
  int wn = (wave >> 2) * 2;
  int m0 = blockIdx.y * 64;
  int n0 = blockIdx.x * 64;
  int lr = tid >> 2;            // 0..63 (tile row)
  int lc = (tid & 3) * 8;       // 0,8,16,24 (k chunk of 8 halves)
  v8f acc0 = {}, acc1 = {};
  for (int k0 = 0; k0 < K; k0 += 32) {
    __syncthreads();
    *(int4*)&sA[lr][lc] = *(const int4*)(A + (size_t)(m0 + lr) * K + k0 + lc);
    *(int4*)&sB[lr][lc] = *(const int4*)(W + (size_t)(n0 + lr) * K + k0 + lc);
    __syncthreads();
    v16h af = load_a(&sA[wm * 16][0], 40);
    v16h b0 = load_b_nk(&sB[wn * 16][0], 40);
    v16h b1 = load_b_nk(&sB[(wn + 1) * 16][0], 40);
    acc0 = wmma16(af, b0, acc0);
    acc1 = wmma16(af, b1, acc1);
  }
  int l = lane_id();
  int ncol = l & 15;
  int mrow = (l >> 4) * 8;
#pragma unroll
  for (int p = 0; p < 2; ++p) {
    v8f ac = p ? acc1 : acc0;
    int ng = n0 + (wn + p) * 16 + ncol;
    float bv = bias ? bias[ng] : 0.f;
#pragma unroll
    for (int j = 0; j < 8; ++j) {
      int mg = m0 + wm * 16 + mrow + j;
      C[(size_t)mg * N + ng] = ac[j] + bv;
    }
  }
}

// ---------------- Kernel 4: per-head LN of q -> [B,H,S,DH] f16 ----------------
__global__ __launch_bounds__(256) void ln_head_q_kernel(
    const float* __restrict__ q, const float* __restrict__ w,
    const float* __restrict__ bsh, _Float16* __restrict__ qo) {
  int tok = blockIdx.x;
  int b = tok / S_, s = tok % S_;
  int h = threadIdx.x >> 5, l = lane_id();
  const float* src = q + (size_t)tok * D_ + h * DH_;
  float e0 = src[l], e1 = src[l + 32];
  float mean = wave_allred(e0 + e1) * (1.f / (float)DH_);
  float d0 = e0 - mean, d1 = e1 - mean;
  float var = wave_allred(d0 * d0 + d1 * d1) * (1.f / (float)DH_);
  float rstd = rsqrtf(var + LN_EPS_);
  size_t o = (((size_t)b * H_ + h) * S_ + s) * DH_;
  qo[o + l]      = (_Float16)(d0 * rstd * w[l] + bsh[l]);
  qo[o + l + 32] = (_Float16)(d1 * rstd * w[l + 32] + bsh[l + 32]);
}

// -------- Kernel 5: per-head LN of k, v->f16, kscale from raw k[...,0] --------
__global__ __launch_bounds__(256) void ln_head_kv_kernel(
    const float* __restrict__ kv, const float* __restrict__ w,
    const float* __restrict__ bsh, const float* __restrict__ zero,
    _Float16* __restrict__ ko, _Float16* __restrict__ vo,
    float* __restrict__ ksc) {
  int tok = blockIdx.x;
  int b = tok / S_, s = tok % S_;
  int h = threadIdx.x >> 5, l = lane_id();
  const float* srck = kv + (size_t)tok * (2 * D_) + h * DH_;
  const float* srcv = srck + D_;
  float e0 = srck[l], e1 = srck[l + 32];
  float mean = wave_allred(e0 + e1) * (1.f / (float)DH_);
  float d0 = e0 - mean, d1 = e1 - mean;
  float var = wave_allred(d0 * d0 + d1 * d1) * (1.f / (float)DH_);
  float rstd = rsqrtf(var + LN_EPS_);
  size_t o = (((size_t)b * H_ + h) * S_ + s) * DH_;
  ko[o + l]      = (_Float16)(d0 * rstd * w[l] + bsh[l]);
  ko[o + l + 32] = (_Float16)(d1 * rstd * w[l + 32] + bsh[l + 32]);
  vo[o + l]      = (_Float16)srcv[l];
  vo[o + l + 32] = (_Float16)srcv[l + 32];
  if (l == 0) {
    float sp = log1pf(__expf(zero[0]));
    float zc = fminf(1e-5f, fmaxf(1e-6f, sp));
    ksc[((size_t)b * H_ + h) * S_ + s] = (e0 == 0.f) ? zc : 1.f;
  }
}

// ---------------- Kernel 6: fused streaming attention (TDM double-buffered) --
// grid = (S/64, B*H); block = 256 (8 waves). Per block: 64 q rows, sweep all
// 32 key tiles of 64 (masked keys still contribute t = 1, per the reference).
// Wave 0 drives the Tensor Data Mover: K/V tile it+1 DMAs into the alternate
// LDS buffer while all waves run WMMA on tile it; TENSORcnt gates handoff.
__global__ __launch_bounds__(256) void attn_kernel(
    const _Float16* __restrict__ qln, const _Float16* __restrict__ kln,
    const _Float16* __restrict__ vh, const float* __restrict__ kscale,
    _Float16* __restrict__ outp) {
  __shared__ _Float16 sQ[64][72];
  __shared__ _Float16 sK[2][64][72];
  __shared__ _Float16 sV[2][64][72];
  __shared__ _Float16 sP[64][72];
  __shared__ float sKs[64];
  __shared__ float sDen[64];

  constexpr int NIT = S_ / 64;          // 32 key tiles
  int bh = blockIdx.y;                  // b*H + h
  int q0 = blockIdx.x * 64;
  size_t base = (size_t)bh * S_ * DH_;
  int tid = threadIdx.x;
  int wave = tid >> 5;
  int wm = wave & 3;
  int wn = (wave >> 2) * 2;
  int l = lane_id();
  int ncol = l & 15;
  int mrow = (l >> 4) * 8;
  int lr = tid >> 2;                    // 0..63
  int lc = (tid & 3) * 16;              // 0,16,32,48

  {
    const _Float16* qs = qln + base + (size_t)(q0 + lr) * DH_ + lc;
    *(int4*)&sQ[lr][lc]     = *(const int4*)qs;
    *(int4*)&sQ[lr][lc + 8] = *(const int4*)(qs + 8);
  }
  if (wave == 0) {                      // prologue: DMA tile 0 into buffer 0
    tdm_load_tile64(kln + base, &sK[0][0][0]);
    tdm_load_tile64(vh + base, &sV[0][0][0]);
  }

  v8f num0 = {}, num1 = {};
  float denreg = 0.f;

  for (int it = 0; it < NIT; ++it) {
    int cur = it & 1, nxt = cur ^ 1;
    int k0 = it * 64;
    __syncthreads();  // readers of buffer `nxt` (iteration it-1) are done
    if (wave == 0) {
      if (it + 1 < NIT) {
        const size_t nb = base + (size_t)(k0 + 64) * DH_;
        tdm_load_tile64(kln + nb, &sK[nxt][0][0]);
        tdm_load_tile64(vh + nb, &sV[nxt][0][0]);
        __builtin_amdgcn_s_wait_tensorcnt(2);  // tile `it` landed; next in flight
      } else {
        __builtin_amdgcn_s_wait_tensorcnt(0);
      }
    }
    if (tid < 64) sKs[tid] = kscale[(size_t)bh * S_ + k0 + tid];
    __syncthreads();  // publish buffer `cur` + sKs to all waves

    // S = Q K^T (each wave: 2 subtiles, Dh split into 2 k-chunks of 32)
    v8f s0 = {}, s1 = {};
#pragma unroll
    for (int c = 0; c < 2; ++c) {
      v16h af = load_a(&sQ[wm * 16][c * 32], 72);
      v16h b0 = load_b_nk(&sK[cur][wn * 16][c * 32], 72);
      v16h b1 = load_b_nk(&sK[cur][(wn + 1) * 16][c * 32], 72);
      s0 = wmma16(af, b0, s0);
      s1 = wmma16(af, b1, s1);
    }

    // mask + sigmoid + Taylor term, write t tile (f16) for the P@V WMMA
#pragma unroll
    for (int p = 0; p < 2; ++p) {
      v8f sc = p ? s1 : s0;
      int nl = (wn + p) * 16 + ncol;
      float ks = sKs[nl];
      int kg = k0 + nl;
#pragma unroll
      for (int j = 0; j < 8; ++j) {
        int ml = wm * 16 + mrow + j;
        float sv = sc[j] * SCALEF_;
        if (kg > q0 + ml) sv = -3.402823466e38f;   // causal: -FLT_MAX
        sv *= ks;                                   // pre-sigmoid scale
        float g = 1.f / (1.f + __expf(-sv));        // sigmoid (-inf -> 0)
        float pv = g * ks;                          // post-sigmoid scale
        float t = 1.f + pv + 0.5f * pv * pv;        // Taylor term (masked -> 1)
        sP[ml][nl] = (_Float16)t;
      }
    }
    __syncthreads();

    // denominator row-sums (threads 0..63 own one q row each)
    if (tid < 64) {
      float rs = 0.f;
#pragma unroll
      for (int c = 0; c < 64; ++c) rs += (float)sP[tid][c];
      denreg += rs;
    }

    // numerator += T @ V (A from sP [m][k], B from sV [k][n])
#pragma unroll
    for (int c = 0; c < 2; ++c) {
      v16h ap = load_a(&sP[wm * 16][c * 32], 72);
      v16h b0 = load_b_kn(&sV[cur][c * 32][wn * 16], 72);
      v16h b1 = load_b_kn(&sV[cur][c * 32][(wn + 1) * 16], 72);
      num0 = wmma16(ap, b0, num0);
      num1 = wmma16(ap, b1, num1);
    }
  }

  __syncthreads();
  if (tid < 64) sDen[tid] = denreg;
  __syncthreads();

  int b = bh >> 3, h = bh & 7;
#pragma unroll
  for (int p = 0; p < 2; ++p) {
    v8f nm = p ? num1 : num0;
    int dl = (wn + p) * 16 + ncol;
#pragma unroll
    for (int j = 0; j < 8; ++j) {
      int ml = wm * 16 + mrow + j;
      float val = nm[j] / sDen[ml];
      outp[((size_t)b * S_ + q0 + ml) * D_ + h * DH_ + dl] = (_Float16)val;
    }
  }
}

// ---------------- launch ----------------
extern "C" void kernel_launch(void* const* d_in, const int* in_sizes, int n_in,
                              void* d_out, int out_size, void* d_ws,
                              size_t ws_size, hipStream_t stream) {
  (void)in_sizes; (void)n_in; (void)out_size; (void)ws_size;
  const float* x     = (const float*)d_in[0];
  const float* Wq    = (const float*)d_in[1];
  const float* bq    = (const float*)d_in[2];
  const float* Wkv   = (const float*)d_in[3];
  const float* Wout  = (const float*)d_in[4];
  const float* bout  = (const float*)d_in[5];
  const float* lna_w = (const float*)d_in[6];
  const float* lna_b = (const float*)d_in[7];
  const float* lnb_w = (const float*)d_in[8];
  const float* lnb_b = (const float*)d_in[9];
  const float* zero  = (const float*)d_in[10];
  // d_in[11] = mask (static 1): causal path compiled in.

  char* ws = (char*)d_ws;
  size_t off = 0;
  auto carve = [&](size_t bytes) {
    void* p = ws + off;
    off += (bytes + 255) & ~(size_t)255;
    return p;
  };
  _Float16* x16   = (_Float16*)carve((size_t)NT_ * D_ * 2);
  _Float16* xln16 = (_Float16*)carve((size_t)NT_ * D_ * 2);
  _Float16* wq16  = (_Float16*)carve((size_t)D_ * D_ * 2);
  _Float16* wkv16 = (_Float16*)carve((size_t)2 * D_ * D_ * 2);
  _Float16* wo16  = (_Float16*)carve((size_t)D_ * D_ * 2);
  float*    qf    = (float*)carve((size_t)NT_ * D_ * 4);
  float*    kvf   = (float*)carve((size_t)NT_ * 2 * D_ * 4);
  _Float16* qln   = (_Float16*)carve((size_t)NT_ * D_ * 2);
  _Float16* kln   = (_Float16*)carve((size_t)NT_ * D_ * 2);
  _Float16* v16   = (_Float16*)carve((size_t)NT_ * D_ * 2);
  float*    ksc   = (float*)carve((size_t)B_ * H_ * S_ * 4);
  _Float16* ao16  = (_Float16*)carve((size_t)NT_ * D_ * 2);

  prep_x_kernel<<<NT_, 256, 0, stream>>>(x, lna_w, lna_b, x16, xln16);
  cvt_f16_kernel<<<256, 256, 0, stream>>>(Wq, wq16, D_ * D_);
  cvt_f16_kernel<<<256, 256, 0, stream>>>(Wkv, wkv16, 2 * D_ * D_);
  cvt_f16_kernel<<<256, 256, 0, stream>>>(Wout, wo16, D_ * D_);

  gemm_wt_kernel<<<dim3(D_ / 64, NT_ / 64), 256, 0, stream>>>(
      x16, wq16, bq, qf, D_, D_);
  gemm_wt_kernel<<<dim3(2 * D_ / 64, NT_ / 64), 256, 0, stream>>>(
      xln16, wkv16, nullptr, kvf, 2 * D_, D_);

  ln_head_q_kernel<<<NT_, 256, 0, stream>>>(qf, lnb_w, lnb_b, qln);
  ln_head_kv_kernel<<<NT_, 256, 0, stream>>>(kvf, lnb_w, lnb_b, zero, kln, v16,
                                             ksc);

  attn_kernel<<<dim3(S_ / 64, B_ * H_), 256, 0, stream>>>(qln, kln, v16, ksc,
                                                          ao16);

  gemm_wt_kernel<<<dim3(D_ / 64, NT_ / 64), 256, 0, stream>>>(
      ao16, wo16, bout, (float*)d_out, D_, D_);
}
